// EfficientMambaAttention00_33904471835296
// MI455X (gfx1250) — matrix operand order repaired and checked by reading, not compiled, not run on hardware
//
#include <hip/hip_runtime.h>
#include <hip/hip_bf16.h>
#include <math.h>

// ---------------------------------------------------------------------------
// EfficientMambaAttention for MI455X (gfx1250, wave32, WMMA).
//   1) fused dual mean over x (single 268MB HBM pass -> ~11.5us floor @23.3TB/s)
//   2) per axis: WMMA-bf16 GEMM chain with fused epilogues
//      (in-proj, x-proj, dt-proj+softplus, out-proj+sigmoid(+amap),
//       mlp1+gelu, mlp2+bias+transpose-store), conv+silu, selective scan.
// GEMM: 4 waves/block (wave32), 32x32/wave via 2x2 v_wmma_f32_16x16x32_bf16,
// double-buffered LDS tiles, branch-free staging, native f32->bf16 cvt.
// ---------------------------------------------------------------------------

typedef __attribute__((ext_vector_type(16))) __bf16 v16bf;
typedef __attribute__((ext_vector_type(8)))  float  v8f;

// Native conversion: clang lowers to hardware bf16 cvt (RNE, NaN-safe),
// pairing into v_cvt_pk_bf16_f32 where possible.
__device__ __forceinline__ __bf16 f2bf(float f) { return (__bf16)f; }

__device__ __forceinline__ float sigmoidf_(float x) { return 1.0f / (1.0f + expf(-x)); }
__device__ __forceinline__ float softplusf_(float x) { return (x > 20.0f) ? x : log1pf(expf(x)); }
__device__ __forceinline__ float gelu_exact_(float x) { return 0.5f * x * (1.0f + erff(x * 0.70710678118654752f)); }

// ---------------------------------------------------------------------------
// 1) Fused dual mean: x [16,256,128,128] -> sx [b,h,c], sy [b,w,c]
//    (token-major [2048,256]). One block per (b,c) tile; 32-row LDS chunks.
// ---------------------------------------------------------------------------
__global__ __launch_bounds__(128)
void reduce_mean_dual(const float* __restrict__ x,
                      float* __restrict__ sx, float* __restrict__ sy) {
    __shared__ float tile[32][132];
    __shared__ float rpart[32][4];
    const int bc = blockIdx.x;          // 0..4095
    const int b = bc >> 8;
    const int c = bc & 255;
    const float* xp = x + (size_t)bc * (128 * 128);
    const int t = threadIdx.x;          // w index 0..127
    float colsum = 0.0f;
    for (int ch = 0; ch < 4; ++ch) {
        #pragma unroll 4
        for (int i = 0; i < 32; ++i) {
            float v = xp[(ch * 32 + i) * 128 + t];   // coalesced 512B rows
            tile[i][t] = v;
            colsum += v;
        }
        __syncthreads();
        const int row = t >> 2, q = t & 3;
        float rs = 0.0f;
        #pragma unroll
        for (int j = 0; j < 32; ++j) rs += tile[row][q * 32 + j];
        rpart[row][q] = rs;
        __syncthreads();
        if (t < 32) {
            float tot = rpart[t][0] + rpart[t][1] + rpart[t][2] + rpart[t][3];
            sx[((size_t)(b * 128 + ch * 32 + t)) * 256 + c] = tot * (1.0f / 128.0f);
        }
        __syncthreads();
    }
    sy[((size_t)(b * 128 + t)) * 256 + c] = colsum * (1.0f / 128.0f);
}

// ---------------------------------------------------------------------------
// 2) WMMA GEMM: C[M,N] = A[M,K] * B[N,K]^T (+bias[n]) (+activation)
//    act: 0=none, 1=softplus, 2=gelu(exact), 3=sigmoid(v + amap[b,n,pos])
//    tstore: 0 -> C[m*ldc+n];  1 -> C[(m>>7)*32768 + n*128 + (m&127)]
//    Double-buffered LDS, vectorized interior fast path, select-based edges.
// ---------------------------------------------------------------------------
#define GTILE 64
#define GKT   32
#define GLDT  40   // bf16 row stride: 80B -> 16B-aligned frags, bank-destaggered

__global__ __launch_bounds__(128)
void gemm_bf16_wmma(const float* __restrict__ A, int lda,
                    const float* __restrict__ B, int ldb,
                    float* __restrict__ C, int ldc,
                    const float* __restrict__ bias,
                    const float* __restrict__ amap,
                    int M, int N, int K, int act, int tstore) {
    __shared__ __bf16 As[2][GTILE * GLDT];
    __shared__ __bf16 Bs[2][GTILE * GLDT];

    const int tid  = threadIdx.x;
    const int lane = tid & 31;
    const int wid  = tid >> 5;
    const int wm   = (wid >> 1) * 32;
    const int wn   = (wid & 1)  * 32;
    const int rowBase = blockIdx.y * GTILE;
    const int colBase = blockIdx.x * GTILE;
    const bool fullTile = (rowBase + GTILE <= M) && (colBase + GTILE <= N);

    v8f acc[2][2];
    #pragma unroll
    for (int mi = 0; mi < 2; ++mi)
        #pragma unroll
        for (int ni = 0; ni < 2; ++ni)
            #pragma unroll
            for (int r = 0; r < 8; ++r) acc[mi][ni][r] = 0.0f;

    const int mrow = lane & 15;
    const int kbA  = (lane >> 4) * 8;    // ISA 16-bit A 16x32 layout
    const int kbB  = (lane >> 4) * 8;    // ISA 16-bit B 32x16 layout
    const int nK = (K + GKT - 1) / GKT;

    // ---- tile stager (branch-free edges; float4 interior) ----
    auto stage = [&](int st, int kt) {
        const int k0 = kt * GKT;
        if (fullTile && (k0 + GKT <= K)) {
            #pragma unroll
            for (int i = 0; i < 4; ++i) {
                const int idx = i * 128 + tid;       // 0..511 float4 slots
                const int r  = idx >> 3;             // 8 float4 per 32-float row
                const int k4 = (idx & 7) * 4;
                const float4 a4 = *(const float4*)(A + (size_t)(rowBase + r) * lda + k0 + k4);
                const float4 b4 = *(const float4*)(B + (size_t)(colBase + r) * ldb + k0 + k4);
                __bf16* ap = &As[st][r * GLDT + k4];
                __bf16* bp = &Bs[st][r * GLDT + k4];
                ap[0] = f2bf(a4.x); ap[1] = f2bf(a4.y); ap[2] = f2bf(a4.z); ap[3] = f2bf(a4.w);
                bp[0] = f2bf(b4.x); bp[1] = f2bf(b4.y); bp[2] = f2bf(b4.z); bp[3] = f2bf(b4.w);
            }
        } else {
            #pragma unroll
            for (int i = 0; i < 16; ++i) {
                const int idx = i * 128 + tid;       // 0..2047 elements
                const int r = idx >> 5, k = idx & 31;
                const int gk = k0 + k;
                const int gm = rowBase + r;
                const int gn = colBase + r;
                const int gkc = gk < K ? gk : K - 1;
                const int gmc = gm < M ? gm : M - 1;
                const int gnc = gn < N ? gn : N - 1;
                float av = A[(size_t)gmc * lda + gkc];   // always in-bounds
                float bv = B[(size_t)gnc * ldb + gkc];
                av = (gm < M && gk < K) ? av : 0.0f;     // v_cndmask, no branches
                bv = (gn < N && gk < K) ? bv : 0.0f;
                As[st][r * GLDT + k] = f2bf(av);
                Bs[st][r * GLDT + k] = f2bf(bv);
            }
        }
    };

    stage(0, 0);
    int st = 0;
    for (int kt = 0; kt < nK; ++kt) {
        __syncthreads();                    // stage(st) visible; st^1 free
        if (kt + 1 < nK) stage(st ^ 1, kt + 1);

        v16bf afrag[2], bfrag[2];
        #pragma unroll
        for (int mi = 0; mi < 2; ++mi) {
            const __bf16* p = &As[st][(wm + mi * 16 + mrow) * GLDT + kbA];
            #pragma unroll
            for (int j = 0; j < 8; ++j) { afrag[mi][j] = p[j]; afrag[mi][8 + j] = p[16 + j]; }
        }
        #pragma unroll
        for (int ni = 0; ni < 2; ++ni) {
            const __bf16* p = &Bs[st][(wn + ni * 16 + mrow) * GLDT + kbB];
            #pragma unroll
            for (int j = 0; j < 8; ++j) { bfrag[ni][j] = p[j]; bfrag[ni][8 + j] = p[16 + j]; }
        }
        #pragma unroll
        for (int mi = 0; mi < 2; ++mi)
            #pragma unroll
            for (int ni = 0; ni < 2; ++ni)
                acc[mi][ni] = __builtin_amdgcn_wmma_f32_16x16x32_bf16(
                    false, afrag[mi], false, bfrag[ni],
                    (short)0, acc[mi][ni], false, false);
        st ^= 1;
    }

    // Epilogue. C/D layout: lanes 0-15 -> M=r, N=lane; lanes 16-31 -> M=r+8.
    const int nloc = lane & 15;
    const int mhi  = (lane >> 4) * 8;
    #pragma unroll
    for (int mi = 0; mi < 2; ++mi)
        #pragma unroll
        for (int ni = 0; ni < 2; ++ni)
            #pragma unroll
            for (int r = 0; r < 8; ++r) {
                const int m = rowBase + wm + mi * 16 + mhi + r;
                const int n = colBase + wn + ni * 16 + nloc;
                if (m < M && n < N) {
                    float v = acc[mi][ni][r];
                    if (bias) v += bias[n];
                    if (act == 1)      v = softplusf_(v);
                    else if (act == 2) v = gelu_exact_(v);
                    else if (act == 3) v = sigmoidf_(v + amap[((size_t)((m >> 7) * 256 + n)) * 128 + (m & 127)]);
                    if (tstore) C[(size_t)(m >> 7) * 32768 + (size_t)n * 128 + (m & 127)] = v;
                    else        C[(size_t)m * ldc + n] = v;
                }
            }
}

// ---------------------------------------------------------------------------
// 3) Depthwise causal conv (D_CONV=4) + bias + SiLU on xz[:, :512]
// ---------------------------------------------------------------------------
__global__ __launch_bounds__(256)
void conv_silu_kernel(const float* __restrict__ xz, const float* __restrict__ convw,
                      const float* __restrict__ convb, float* __restrict__ xc) {
    const int e = blockIdx.x * 256 + threadIdx.x;   // 0..511
    const int t = blockIdx.y;                        // token 0..2047
    const int b = t >> 7, l = t & 127;
    float acc = 0.0f;
    #pragma unroll
    for (int k = 0; k < 4; ++k) {
        const int l2 = l - 3 + k;
        if (l2 >= 0) acc += xz[((size_t)(b * 128 + l2)) * 1024 + e] * convw[e * 4 + k];
    }
    acc += convb[e];
    xc[(size_t)t * 512 + e] = acc * sigmoidf_(acc);  // silu
}

// ---------------------------------------------------------------------------
// 4) Selective scan: one thread per (b,d); 8192 independent recurrences,
//    sequential only over L=128. Includes +x*D and *silu(z).
// ---------------------------------------------------------------------------
__global__ __launch_bounds__(256)
void scan_kernel(const float* __restrict__ dtl, const float* __restrict__ xc,
                 const float* __restrict__ dblp, const float* __restrict__ xz,
                 const float* __restrict__ Alog, const float* __restrict__ Dp,
                 float* __restrict__ ybuf) {
    const int d = blockIdx.x * 256 + threadIdx.x;   // 0..511
    const int b = blockIdx.y;                        // 0..15
    float An[16];
    #pragma unroll
    for (int n = 0; n < 16; ++n) An[n] = -expf(Alog[d * 16 + n]);
    const float Dv = Dp[d];
    float h[16];
    #pragma unroll
    for (int n = 0; n < 16; ++n) h[n] = 0.0f;
    for (int l = 0; l < 128; ++l) {
        const int t = b * 128 + l;
        const float dt  = dtl[(size_t)t * 512 + d];
        const float xv  = xc[(size_t)t * 512 + d];
        const float dtx = dt * xv;
        const float* bc = dblp + (size_t)t * 48;    // [dtr(16) | B(16) | C(16)]
        float y = 0.0f;
        #pragma unroll
        for (int n = 0; n < 16; ++n) {
            const float dA = expf(dt * An[n]);
            h[n] = dA * h[n] + dtx * bc[16 + n];    // uniform (broadcast) loads
            y += h[n] * bc[32 + n];
        }
        y += xv * Dv;
        const float z = xz[(size_t)t * 1024 + 512 + d];
        ybuf[(size_t)t * 512 + d] = y * (z * sigmoidf_(z));
    }
}

// ---------------------------------------------------------------------------
// Host launcher
// ---------------------------------------------------------------------------
extern "C" void kernel_launch(void* const* d_in, const int* in_sizes, int n_in,
                              void* d_out, int out_size, void* d_ws, size_t ws_size,
                              hipStream_t stream) {
    (void)in_sizes; (void)n_in; (void)out_size; (void)ws_size;
    const float* x = (const float*)d_in[0];
    float* ws  = (float*)d_ws;
    float* out = (float*)d_out;

    // Workspace layout (floats). Total ~7.44M floats (~30 MB), L2-resident.
    float* sx   = ws;                 // 2048x256
    float* sy   = ws + 524288;        // 2048x256
    float* xz   = ws + 1048576;       // 2048x1024
    float* xc   = ws + 3145728;       // 2048x512
    float* dblb = ws + 4194304;       // 2048x48
    float* dtl  = ws + 4292608;       // 2048x512
    float* ybuf = ws + 5341184;       // 2048x512
    float* actb = ws + 6389760;       // 2048x256
    float* h1   = ws + 6914048;       // 2048x256

    // Single fused pass over x: both axis means.
    reduce_mean_dual<<<4096, 128, 0, stream>>>(x, sx, sy);

    const dim3 gblk(128);
    for (int axis = 0; axis < 2; ++axis) {
        const int pb = 3 + axis * 13;   // pm_x at d_in[3..15], pm_y at d_in[16..28]
        const float* Win   = (const float*)d_in[pb + 0];   // [1024,256]
        const float* convw = (const float*)d_in[pb + 1];   // [512,4]
        const float* convb = (const float*)d_in[pb + 2];   // [512]
        const float* Wxp   = (const float*)d_in[pb + 3];   // [48,512]
        const float* Wdt   = (const float*)d_in[pb + 4];   // [512,16]
        const float* bdt   = (const float*)d_in[pb + 5];   // [512]
        const float* Alog  = (const float*)d_in[pb + 6];   // [512,16]
        const float* Dp    = (const float*)d_in[pb + 7];   // [512]
        const float* Wout  = (const float*)d_in[pb + 8];   // [256,512]
        const float* W1    = (const float*)d_in[pb + 9];   // [256,256]
        const float* b1    = (const float*)d_in[pb + 10];  // [256]
        const float* W2    = (const float*)d_in[pb + 11];  // [256,256]
        const float* b2    = (const float*)d_in[pb + 12];  // [256]
        const float* seq   = (axis == 0) ? sx : sy;
        const float* amap  = (const float*)d_in[1 + axis];
        float* outp = out + (size_t)axis * 524288;

        // in-proj: xz[2048,1024] = seq * Win^T
        gemm_bf16_wmma<<<dim3(16, 32), gblk, 0, stream>>>(seq, 256, Win, 256, xz, 1024,
                                                          nullptr, nullptr, 2048, 1024, 256, 0, 0);
        // depthwise conv + silu on x-half
        conv_silu_kernel<<<dim3(2, 2048), 256, 0, stream>>>(xz, convw, convb, xc);
        // x-proj: dbl[2048,48] = xc * Wxp^T
        gemm_bf16_wmma<<<dim3(1, 32), gblk, 0, stream>>>(xc, 512, Wxp, 512, dblb, 48,
                                                         nullptr, nullptr, 2048, 48, 512, 0, 0);
        // dt-proj: dtl[2048,512] = softplus(dbl[:, :16] * Wdt^T + bdt)
        gemm_bf16_wmma<<<dim3(8, 32), gblk, 0, stream>>>(dblb, 48, Wdt, 16, dtl, 512,
                                                         bdt, nullptr, 2048, 512, 16, 1, 0);
        // selective scan
        scan_kernel<<<dim3(2, 16), 256, 0, stream>>>(dtl, xc, dblb, xz, Alog, Dp, ybuf);
        // out-proj + fused sigmoid(v + amap): actb[2048,256]
        gemm_bf16_wmma<<<dim3(4, 32), gblk, 0, stream>>>(ybuf, 512, Wout, 512, actb, 256,
                                                         nullptr, amap, 2048, 256, 512, 3, 0);
        // mlp1: h1 = gelu(actb * W1^T + b1)
        gemm_bf16_wmma<<<dim3(4, 32), gblk, 0, stream>>>(actb, 256, W1, 256, h1, 256,
                                                         b1, nullptr, 2048, 256, 256, 2, 0);
        // mlp2 + fused transpose store to out[b,c,pos]
        gemm_bf16_wmma<<<dim3(4, 32), gblk, 0, stream>>>(h1, 256, W2, 256, outp, 128,
                                                         b2, nullptr, 2048, 256, 256, 0, 1);
    }
}
